// EV_GCN_65463891525765
// MI455X (gfx1250) — compile-verified, matrix operand order checked by reading, not run on hardware
//
#include <hip/hip_runtime.h>
#include <math.h>

#define Nn   20000
#define Ee   320000
#define Pp   8192

typedef __attribute__((ext_vector_type(16))) _Float16 v16h;
typedef __attribute__((ext_vector_type(8)))  float    v8f;

// ---------------------------------------------------------------------------
// WMMA helpers (CDNA5 wave32, V_WMMA_F32_16X16X32_F16)
// A fragment: lane l<16 -> row (l&15), K in {kb..kb+7, 16+kb..16+kb+7}, kb=8*(l>>4)
// B fragment: same pattern with row == output column N (weights stored N-major,K-contig)
// C/D: lane l -> col N=(l&15); vgpr r -> row M = 8*(l>>4)+r
// ---------------------------------------------------------------------------
__device__ __forceinline__ v16h frag16(const _Float16* __restrict__ p, int ld, int lane) {
  int r  = lane & 15;
  int kb = (lane >> 4) << 3;
  const _Float16* q = p + r * ld + kb;
  v16h f;
#pragma unroll
  for (int i = 0; i < 8; ++i) { f[i] = q[i]; f[i + 8] = q[i + 16]; }
  return f;
}

__device__ __forceinline__ v8f wmma32(v16h a, v16h b, v8f c) {
  return __builtin_amdgcn_wmma_f32_16x16x32_f16(false, a, false, b, (short)0, c,
                                                false, false);
}

// ---------------------------------------------------------------------------
// Fused PAE: per block -> 64 edges (128 endpoint rows).
// h = relu(x@W1^T+b1); h2 = h@W2^T+b2; cos(h2_a,h2_b) -> edge weight in [0,1]
// ---------------------------------------------------------------------------
__global__ __launch_bounds__(256) void k_pae(
    const float* __restrict__ enet,   // [E,32]
    const float* __restrict__ w1,     // [128,16]
    const float* __restrict__ b1,     // [128]
    const float* __restrict__ w2,     // [128,128]
    const float* __restrict__ b2,     // [128]
    float* __restrict__ wE, int E) {
  __shared__ _Float16 sW1[128 * 32];    // [n][k] K padded 16->32 with zeros
  __shared__ _Float16 sW2[128 * 136];   // [n][k]
  __shared__ float    sB1[128], sB2[128];
  __shared__ _Float16 sX [128 * 40];    // [row][k] K padded 16->32
  __shared__ _Float16 sH1[128 * 136];
  __shared__ _Float16 sH2[128 * 136];

  int tid  = threadIdx.x;
  int lane = tid & 31;
  int wid  = tid >> 5;
  int e0   = blockIdx.x * 64;

  for (int i = tid; i < 128 * 32; i += 256) {
    int n = i >> 5, k = i & 31;
    sW1[i] = (k < 16) ? (_Float16)w1[n * 16 + k] : (_Float16)0.f;
  }
  for (int i = tid; i < 128 * 128; i += 256) {
    int n = i >> 7, k = i & 127;
    sW2[n * 136 + k] = (_Float16)w2[i];
  }
  if (tid < 128) { sB1[tid] = b1[tid]; sB2[tid] = b2[tid]; }
  // rows 0..63: endpoint 0 of edges e0..e0+63; rows 64..127: endpoint 1
  for (int i = tid; i < 128 * 32; i += 256) {
    int r = i >> 5, k = i & 31;
    int e = e0 + (r & 63), ep = r >> 6;
    float v = 0.f;
    if (k < 16 && e < E) v = enet[e * 32 + ep * 16 + k];
    sX[r * 40 + k] = (_Float16)v;
  }
  __syncthreads();

  v8f acc[8] = {};
  // GEMM1: 128x128 = X(128x32) x W1^T, single K step
  {
    v16h a = frag16(sX + wid * 16 * 40, 40, lane);
#pragma unroll
    for (int ns = 0; ns < 8; ++ns) {
      v16h b = frag16(sW1 + ns * 16 * 32, 32, lane);
      acc[ns] = wmma32(a, b, acc[ns]);
    }
    int n = lane & 15, mr = (lane >> 4) * 8;
#pragma unroll
    for (int ns = 0; ns < 8; ++ns) {
      int   gn = ns * 16 + n;
      float bb = sB1[gn];
#pragma unroll
      for (int r = 0; r < 8; ++r) {
        float v = acc[ns][r] + bb;
        v = v > 0.f ? v : 0.f;
        sH1[(wid * 16 + mr + r) * 136 + gn] = (_Float16)v;
      }
    }
  }
  __syncthreads();

  // GEMM2: 128x128 = H1(128x128) x W2^T, 4 K steps
  {
    v8f vz = {};
#pragma unroll
    for (int ns = 0; ns < 8; ++ns) acc[ns] = vz;
  }
  for (int kc = 0; kc < 4; ++kc) {
    v16h a = frag16(sH1 + wid * 16 * 136 + kc * 32, 136, lane);
#pragma unroll
    for (int ns = 0; ns < 8; ++ns) {
      v16h b = frag16(sW2 + ns * 16 * 136 + kc * 32, 136, lane);
      acc[ns] = wmma32(a, b, acc[ns]);
    }
  }
  {
    int n = lane & 15, mr = (lane >> 4) * 8;
#pragma unroll
    for (int ns = 0; ns < 8; ++ns) {
      int   gn = ns * 16 + n;
      float bb = sB2[gn];
#pragma unroll
      for (int r = 0; r < 8; ++r)
        sH2[(wid * 16 + mr + r) * 136 + gn] = (_Float16)(acc[ns][r] + bb);
    }
  }
  __syncthreads();

  if (tid < 64) {
    int e = e0 + tid;
    if (e < E) {
      float dot = 0.f, n1 = 0.f, n2 = 0.f;
      for (int c = 0; c < 128; ++c) {
        float a = (float)sH2[tid * 136 + c];
        float b = (float)sH2[(64 + tid) * 136 + c];
        dot += a * b; n1 += a * a; n2 += b * b;
      }
      float cosv = dot / (sqrtf(n1) * sqrtf(n2) + 1e-8f);
      wE[e] = 0.5f * (cosv + 1.f);
    }
  }
}

// ---------------------------------------------------------------------------
// Generic WMMA GEMM:  D = act( A[M,K] x B + bias (+ D) )
// wt==1: W is [Ntot,Ktot] (torch Linear, ldw=Ktot) ; wt==0: W is [Ktot,Ntot] (ldw=Ntot)
// Tile: BM=128, BN=64, BK=32 ; 256 threads = 8 waves, 4 accumulators each
// ---------------------------------------------------------------------------
__global__ __launch_bounds__(256) void k_wmma_gemm(
    const float* __restrict__ A, int lda,
    const float* __restrict__ W, int ldw, int wt,
    const float* __restrict__ bias,
    float* __restrict__ D, int ldd,
    int M, int Ntot, int Ktot, int accum, int relu) {
  __shared__ _Float16 sA[128 * 40];
  __shared__ _Float16 sB[64 * 40];
  int tid  = threadIdx.x;
  int lane = tid & 31;
  int wid  = tid >> 5;
  int m0   = blockIdx.x * 128;
  int n0   = blockIdx.y * 64;

  v8f acc[4] = {};
  for (int k0 = 0; k0 < Ktot; k0 += 32) {
    __syncthreads();
    for (int i = tid; i < 128 * 32; i += 256) {
      int r = i >> 5, k = i & 31;
      int gm = m0 + r;
      float v = (gm < M) ? A[(size_t)gm * lda + k0 + k] : 0.f;
      sA[r * 40 + k] = (_Float16)v;
    }
    for (int i = tid; i < 64 * 32; i += 256) {
      int n = i >> 5, k = i & 31;
      int gn = n0 + n, gk = k0 + k;
      float v = 0.f;
      if (gn < Ntot)
        v = wt ? W[(size_t)gn * ldw + gk] : W[(size_t)gk * ldw + gn];
      sB[n * 40 + k] = (_Float16)v;
    }
    __syncthreads();
    v16h a = frag16(sA + wid * 16 * 40, 40, lane);
#pragma unroll
    for (int ns = 0; ns < 4; ++ns) {
      v16h b = frag16(sB + ns * 16 * 40, 40, lane);
      acc[ns] = wmma32(a, b, acc[ns]);
    }
  }

  int n = lane & 15, mr = (lane >> 4) * 8;
#pragma unroll
  for (int ns = 0; ns < 4; ++ns) {
    int   gn = n0 + ns * 16 + n;
    float bv = bias ? bias[gn] : 0.f;
#pragma unroll
    for (int r = 0; r < 8; ++r) {
      int gm = m0 + wid * 16 + mr + r;
      if (gm < M) {
        float v = acc[ns][r] + bv;
        if (accum) v += D[(size_t)gm * ldd + gn];
        if (relu)  v = v > 0.f ? v : 0.f;
        D[(size_t)gm * ldd + gn] = v;
      }
    }
  }
}

// ---------------------------------------------------------------------------
// Elementwise / graph kernels
// ---------------------------------------------------------------------------
__global__ void k_zero(float* __restrict__ p, int n) {
  int i = blockIdx.x * blockDim.x + threadIdx.x;
  if (i < n) p[i] = 0.f;
}

// wfull = w - w*m ; norms ; ew = threshold(wfull) ; deg[row] += ew
__global__ void k_graph1(const float* __restrict__ wE, const float* __restrict__ m,
                         const int* __restrict__ row, float* __restrict__ ew,
                         float* __restrict__ deg, float* __restrict__ s, int E) {
  float sw = 0.f, sm = 0.f;
  for (int i = blockIdx.x * blockDim.x + threadIdx.x; i < E;
       i += blockDim.x * gridDim.x) {
    float mv = m[i];
    float wf = wE[i] * (1.f - mv);
    sw += wf * wf; sm += mv * mv;
    float t = wf > 0.25f ? wf : 0.f;
    ew[i] = t;
    if (t > 0.f) atomicAdd(&deg[row[i]], t);
  }
  atomicAdd(&s[0], sw);
  atomicAdd(&s[1], sm);
}

__global__ void k_graph2(const float* __restrict__ deg, float* __restrict__ dinv,
                         float* __restrict__ diag, int N) {
  int i = blockIdx.x * blockDim.x + threadIdx.x;
  if (i < N) {
    float d = deg[i];
    dinv[i] = d > 0.f ? rsqrtf(fmaxf(d, 1e-12f)) : 0.f;
    diag[i] = d > 0.f ? 0.f : -1.f;
  }
}

__global__ void k_graph3(const int* __restrict__ row, const int* __restrict__ col,
                         const float* __restrict__ ew, const float* __restrict__ dinv,
                         float* __restrict__ lapw, int E) {
  int i = blockIdx.x * blockDim.x + threadIdx.x;
  if (i < E) lapw[i] = -dinv[row[i]] * ew[i] * dinv[col[i]];
}

// r = alpha*diag*t (+ gamma*tprev)
__global__ void k_prop_init(const float* __restrict__ t, const float* __restrict__ tprev,
                            const float* __restrict__ diag, float* __restrict__ r,
                            int total, int logF, float alpha, float gamma) {
  int i = blockIdx.x * blockDim.x + threadIdx.x;
  if (i < total) {
    float v = alpha * diag[i >> logF] * t[i];
    if (tprev) v += gamma * tprev[i];
    r[i] = v;
  }
}

// r[col[e]] += alpha*lapw[e]*t[row[e]]  (float4 per thread, atomic f32)
__global__ void k_prop_scatter(const int* __restrict__ row, const int* __restrict__ col,
                               const float* __restrict__ lapw, const float* __restrict__ t,
                               float* __restrict__ r, int E, int F, int lp, float alpha) {
  int tid = blockIdx.x * blockDim.x + threadIdx.x;
  int e = tid >> lp;
  int c = (tid & ((1 << lp) - 1)) << 2;
  if (e < E) {
    __builtin_prefetch(lapw + e + 2048, 0, 0);  // global_prefetch_b8
    float wv = alpha * lapw[e];
    if (wv != 0.f) {
      const float4 tv = *reinterpret_cast<const float4*>(t + (size_t)row[e] * F + c);
      float* dst = r + (size_t)col[e] * F + c;
      atomicAdd(dst + 0, wv * tv.x);
      atomicAdd(dst + 1, wv * tv.y);
      atomicAdd(dst + 2, wv * tv.z);
      atomicAdd(dst + 3, wv * tv.w);
    }
  }
}

__global__ void k_relu_concat(const float* __restrict__ acc, float* __restrict__ h,
                              float* __restrict__ h0, int off, int N) {
  int i = blockIdx.x * blockDim.x + threadIdx.x;
  if (i < N * 64) {
    int n = i >> 6, j = i & 63;
    float v = fmaxf(acc[i], 0.f);
    h[i] = v;
    h0[n * 256 + off + j] = v;
  }
}

__global__ void k_cos_exp(const float* __restrict__ hd, const float* __restrict__ hh,
                          float* __restrict__ c, int N) {
  int n = blockIdx.x * blockDim.x + threadIdx.x;
  if (n < N) {
    float dot = 0.f, n1 = 0.f, n2 = 0.f;
    for (int j = 0; j < 64; ++j) {
      float a = hd[n * 64 + j], b = hh[n * 64 + j];
      dot += a * b; n1 += a * a; n2 += b * b;
    }
    float cosv = dot / (sqrtf(n1) * sqrtf(n2) + 1e-8f);
    c[n] = expf(cosv * 2.0f);  // / TQ (=0.5)
  }
}

__global__ void k_gather_sum(const float* __restrict__ c, const int* __restrict__ idx,
                             int P, float* __restrict__ s) {
  float a = 0.f;
  for (int i = blockIdx.x * blockDim.x + threadIdx.x; i < P;
       i += blockDim.x * gridDim.x)
    a += c[idx[i]];
  atomicAdd(s, a);
}

__global__ void k_sumsq(const float* __restrict__ x, int n, float* __restrict__ s) {
  float a = 0.f;
  for (int i = blockIdx.x * blockDim.x + threadIdx.x; i < n;
       i += blockDim.x * gridDim.x)
    a += x[i] * x[i];
  atomicAdd(s, a);
}

__global__ void k_go(const float* __restrict__ hd, const float* __restrict__ hh,
                     float* __restrict__ go, int N) {
  int i = blockIdx.x * blockDim.x + threadIdx.x;
  if (i < N * 128) {
    int n = i >> 7, j = i & 127;
    go[i] = (j < 64) ? hd[n * 64 + j] : (hd[n * 64 + j - 64] - hh[n * 64 + j - 64]);
  }
}

__global__ void k_cls_out(const float* __restrict__ z, const float* __restrict__ w2,
                          const float* __restrict__ b2, float* __restrict__ out, int N) {
  int i = blockIdx.x * blockDim.x + threadIdx.x;
  if (i < N * 2) {
    int n = i >> 1, cl = i & 1;
    float d = b2[cl];
    for (int j = 0; j < 256; ++j) d += z[n * 256 + j] * w2[cl * 256 + j];
    out[i] = d;
  }
}

__global__ void k_final(const float* __restrict__ s, float* __restrict__ out, int N) {
  if (threadIdx.x == 0 && blockIdx.x == 0) {
    out[2 * N + 0] = sqrtf(s[0]) + sqrtf(s[1]);   // norm
    out[2 * N + 1] = -logf(s[2] / s[3]);          // loss2
    out[2 * N + 2] = sqrtf(s[4]) + sqrtf(s[5]);   // loss1
  }
}

// ---------------------------------------------------------------------------
extern "C" void kernel_launch(void* const* d_in, const int* in_sizes, int n_in,
                              void* d_out, int out_size, void* d_ws, size_t ws_size,
                              hipStream_t stream) {
  (void)in_sizes; (void)n_in; (void)out_size; (void)ws_size;
  const float* feat_dis = (const float*)d_in[0];
  const int*   eidx     = (const int*)d_in[1];
  const float* enet     = (const float*)d_in[2];
  const float* feat_hea = (const float*)d_in[3];
  const int*   post_ind = (const int*)d_in[6];
  const int*   nega_ind = (const int*)d_in[7];
  const float* pae_w1 = (const float*)d_in[8];
  const float* pae_b1 = (const float*)d_in[9];
  const float* pae_w2 = (const float*)d_in[10];
  const float* pae_b2 = (const float*)d_in[11];
  const float* cheb_w[4] = {(const float*)d_in[12], (const float*)d_in[13],
                            (const float*)d_in[14], (const float*)d_in[15]};
  const float* matrix = (const float*)d_in[16];
  const float* ed_w1 = (const float*)d_in[17];
  const float* ed_b1 = (const float*)d_in[18];
  const float* ed_w2 = (const float*)d_in[19];
  const float* ed_b2 = (const float*)d_in[20];
  const float* eh_w1 = (const float*)d_in[21];
  const float* eh_b1 = (const float*)d_in[22];
  const float* eh_w2 = (const float*)d_in[23];
  const float* eh_b2 = (const float*)d_in[24];
  const float* cls_w1 = (const float*)d_in[25];
  const float* cls_b1 = (const float*)d_in[26];
  const float* cls_w2 = (const float*)d_in[27];
  const float* cls_b2 = (const float*)d_in[28];
  float* out = (float*)d_out;

  const int* row = eidx;
  const int* col = eidx + Ee;

  // workspace carve-up (floats)
  float* p    = (float*)d_ws;
  float* wE   = p; p += Ee;
  float* ew   = p; p += Ee;
  float* lapw = p; p += Ee;
  float* deg  = p; p += Nn;
  float* dinv = p; p += Nn;
  float* diag = p; p += Nn;
  float* Bu1  = p; p += Nn * 128;
  float* Bu2  = p; p += Nn * 128;
  float* Bu3  = p; p += Nn * 128;
  float* acc  = p; p += Nn * 64;
  float* h    = p; p += Nn * 64;
  float* h0   = p; p += Nn * 256;
  float* z1   = p; p += Nn * 512;
  float* hd   = p; p += Nn * 64;
  float* hh   = p; p += Nn * 64;
  float* go   = p; p += Nn * 128;
  float* cexp = p; p += Nn;
  float* scal = p; p += 16;

  auto gemm = [&](const float* A, int lda, const float* W, int ldw, int wt,
                  const float* bias, float* D, int ldd, int M, int Ntot, int Ktot,
                  int accum, int relu) {
    dim3 g((M + 127) / 128, Ntot / 64);
    k_wmma_gemm<<<g, 256, 0, stream>>>(A, lda, W, ldw, wt, bias, D, ldd, M, Ntot,
                                       Ktot, accum, relu);
  };

  // init
  k_zero<<<(Nn + 255) / 256, 256, 0, stream>>>(deg, Nn);
  k_zero<<<1, 32, 0, stream>>>(scal, 16);

  // PAE -> edge weights
  k_pae<<<Ee / 64, 256, 0, stream>>>(enet, pae_w1, pae_b1, pae_w2, pae_b2, wE, Ee);

  // graph_learn
  k_graph1<<<256, 256, 0, stream>>>(wE, matrix, row, ew, deg, scal, Ee);
  k_graph2<<<(Nn + 255) / 256, 256, 0, stream>>>(deg, dinv, diag, Nn);
  k_graph3<<<(Ee + 255) / 256, 256, 0, stream>>>(row, col, ew, dinv, lapw, Ee);

  // Chebyshev branch (shared across dis/hea)
  auto branch = [&](const float* feat) {
    const float* x = feat;
    for (int l = 0; l < 4; ++l) {
      int Fin  = l ? 64 : 128;
      int logF = l ? 6 : 7;
      int lp   = l ? 4 : 5;  // log2(Fin/4)
      const float* W = cheb_w[l];
      int total = Nn * Fin;
      // k=0 term: acc = x @ w[0]
      gemm(x, Fin, W, 64, 0, nullptr, acc, 64, Nn, 64, Fin, 0, 0);
      // tx1 = prop(x) = scatter + diag*x  -> Bu1
      k_prop_init<<<(total + 255) / 256, 256, 0, stream>>>(x, nullptr, diag, Bu1,
                                                           total, logF, 1.f, 0.f);
      k_prop_scatter<<<(Ee * (Fin / 4)) / 256, 256, 0, stream>>>(row, col, lapw, x,
                                                                 Bu1, Ee, Fin, lp, 1.f);
      gemm(Bu1, Fin, W + 1 * Fin * 64, 64, 0, nullptr, acc, 64, Nn, 64, Fin, 1, 0);
      const float* ta = x; float* tb = Bu1; float* tc = Bu2;
      for (int k = 2; k < 6; ++k) {
        // tx2 = 2*prop(tx1) - tx0
        k_prop_init<<<(total + 255) / 256, 256, 0, stream>>>(tb, ta, diag, tc,
                                                             total, logF, 2.f, -1.f);
        k_prop_scatter<<<(Ee * (Fin / 4)) / 256, 256, 0, stream>>>(row, col, lapw, tb,
                                                                   tc, Ee, Fin, lp, 2.f);
        gemm(tc, Fin, W + k * Fin * 64, 64, 0, nullptr, acc, 64, Nn, 64, Fin, 1, 0);
        float* ntc = (k == 2) ? Bu3 : (float*)ta;
        ta = tb; tb = tc; tc = ntc;
      }
      k_relu_concat<<<(Nn * 64 + 255) / 256, 256, 0, stream>>>(acc, h, h0, l * 64, Nn);
      x = h;
    }
  };

  // dis branch + encoder -> hd
  branch(feat_dis);
  gemm(h0, 256, ed_w1, 256, 1, ed_b1, z1, 512, Nn, 512, 256, 0, 1);
  gemm(z1, 512, ed_w2, 512, 1, ed_b2, hd, 64, Nn, 64, 512, 0, 0);

  // hea branch (reuses dis graph, as in source) + encoder -> hh
  branch(feat_hea);
  gemm(h0, 256, eh_w1, 256, 1, eh_b1, z1, 512, Nn, 512, 256, 0, 1);
  gemm(z1, 512, eh_w2, 512, 1, eh_b2, hh, 64, Nn, 64, 512, 0, 0);

  // contrastive pieces
  k_cos_exp<<<(Nn + 255) / 256, 256, 0, stream>>>(hd, hh, cexp, Nn);
  k_gather_sum<<<32, 256, 0, stream>>>(cexp, post_ind, Pp, &scal[2]);
  k_gather_sum<<<32, 256, 0, stream>>>(cexp, nega_ind, Pp, &scal[3]);
  k_sumsq<<<32, 256, 0, stream>>>(ed_w2, 64 * 512, &scal[4]);
  k_sumsq<<<32, 256, 0, stream>>>(eh_w2, 64 * 512, &scal[5]);

  // classifier: go = [hd, hd-hh] -> relu(go@W1^T+b1) -> logits
  k_go<<<(Nn * 128 + 255) / 256, 256, 0, stream>>>(hd, hh, go, Nn);
  gemm(go, 128, cls_w1, 128, 1, cls_b1, z1, 256, Nn, 256, 128, 0, 1);
  k_cls_out<<<(Nn * 2 + 255) / 256, 256, 0, stream>>>(z1, cls_w2, cls_b2, out, Nn);

  // scalars: norm, loss2, loss1
  k_final<<<1, 32, 0, stream>>>(scal, out, Nn);
}
